// Glm4vMoeTextTopkRouter_86208583565711
// MI455X (gfx1250) — matrix operand order repaired and checked by reading, not compiled, not run on hardware
//
#include <hip/hip_runtime.h>
#include <math.h>

#define HIDDEN 4096
#define NEXP   128
#define TOPK   8

typedef __attribute__((ext_vector_type(16))) __bf16          v16bf;
typedef __attribute__((ext_vector_type(16))) unsigned short  v16us;
typedef __attribute__((ext_vector_type(8)))  float           v8f;

union Frag32B {
    uint4 u4[2];
    v16us us;
    v16bf bf;
};

// Split fp32 into hi/lo bf16 by bit truncation: f = hi + r, lo = trunc_bf16(r).
// Overall representation error ~2^-16 relative, recovered by 3-term WMMA.
__device__ __forceinline__ void bfsplit(float f, unsigned short& h, unsigned short& l) {
    unsigned u = __float_as_uint(f);
    h = (unsigned short)(u >> 16);
    float r = f - __uint_as_float(u & 0xffff0000u);
    l = (unsigned short)(__float_as_uint(r) >> 16);
}

// Pack hi/lo bf16 of 8 floats into one uint4 each, using v_perm_b32 packing.
__device__ __forceinline__ void cvt8(const float4 a, const float4 b,
                                     uint4& hi, uint4& lo) {
    float f[8] = {a.x, a.y, a.z, a.w, b.x, b.y, b.z, b.w};
    unsigned h[4], l[4];
#pragma unroll
    for (int p = 0; p < 4; ++p) {
        unsigned u0 = __float_as_uint(f[2 * p]);
        unsigned u1 = __float_as_uint(f[2 * p + 1]);
        h[p] = __builtin_amdgcn_perm(u1, u0, 0x07060302u);   // {hi16(f1),hi16(f0)}
        float r0 = f[2 * p]     - __uint_as_float(u0 & 0xffff0000u);
        float r1 = f[2 * p + 1] - __uint_as_float(u1 & 0xffff0000u);
        l[p] = __builtin_amdgcn_perm(__float_as_uint(r1), __float_as_uint(r0),
                                     0x07060302u);
    }
    hi = make_uint4(h[0], h[1], h[2], h[3]);
    lo = make_uint4(l[0], l[1], l[2], l[3]);
}

// ---------------------------------------------------------------------------
// Kernel 0: pre-swizzle router weights into B-fragment order, split hi/lo bf16.
// B 32x16 bf16 fragment layout (contiguous k order, same as A): lane L holds column
// N = L&15; lane-half s holds 16 contiguous K = (kc*2+s)*16 + 0..15.
// Storage: Wsw[f][e][j], f = K/16 = 0..HIDDEN/16-1, j = 0..15.
// ---------------------------------------------------------------------------
__global__ __launch_bounds__(256) void preswz_kernel(
    const float* __restrict__ W,
    unsigned short* __restrict__ whi,
    unsigned short* __restrict__ wlo) {
    int t = blockIdx.x * blockDim.x + threadIdx.x;   // 0 .. (HIDDEN/16)*NEXP-1
    int f = t / NEXP;
    int e = t % NEXP;
    const float* src = W + (size_t)e * HIDDEN + (size_t)f * 16;
    int base = t * 16;
#pragma unroll
    for (int j = 0; j < 16; ++j) {
        unsigned short h, l;
        bfsplit(src[j], h, l);
        whi[base + j] = h;
        wlo[base + j] = l;
    }
}

// ---------------------------------------------------------------------------
// Kernel 1: logits[T,128] = x[T,4096] . W^T via bf16-split WMMA.
// Block = 256 threads = 8 waves; block computes 64 tokens x 128 experts.
// Wave w owns experts [16w,16w+16) and FOUR 16-token M-tiles (4 accumulators),
// so each B fragment is reused 12x (4 M-tiles x 3 split terms).
// The cooperative loader converts fp32 -> packed hi/lo bf16 ONCE into LDS;
// waves read ready A fragments with ds_load_b128 (no per-wave conversion).
// ---------------------------------------------------------------------------
#define MBLK 64
#define BK   128
#define STR  (BK + 8)            // bf16 row stride: 272B = 68 dwords -> no conflicts
#define LOFF (MBLK * STR)        // Alo tile offset (ushort elements)

__global__ __launch_bounds__(256) void router_gemm_kernel(
    const float* __restrict__ X,
    const unsigned short* __restrict__ Whi,
    const unsigned short* __restrict__ Wlo,
    float* __restrict__ logits) {
    __shared__ unsigned short axs[2 * MBLK * STR];   // Ahi | Alo, 34816 B

    const int tid  = threadIdx.x;
    const int wave = tid >> 5;
    const int lane = tid & 31;
    const int r    = lane & 15;     // row/col within 16x16 tile
    const int s    = lane >> 4;     // lane-half selector
    const int n0   = wave * 16;     // expert tile base
    const int tokenBase = blockIdx.x * MBLK;

    v8f acc[4] = {};

    // loader mapping: 1024 8-float chunks = 64 rows x 16 groups; 4 per thread
    const int row0 = tid >> 4;           // chunk c = tid + 256*i -> row = c>>4
    const int cg   = tid & 15;           // same column group for all 4 chunks

    for (int kt = 0; kt < HIDDEN / BK; ++kt) {
        // ---- stage x tile: global fp32 -> registers -> split -> LDS bf16
        float4 g[4][2];
#pragma unroll
        for (int i = 0; i < 4; ++i) {
            const float4* src4 = (const float4*)(X +
                (size_t)(tokenBase + row0 + 16 * i) * HIDDEN + kt * BK + cg * 8);
            g[i][0] = src4[0];
            g[i][1] = src4[1];
        }
        __syncthreads();                 // previous stage reads done
#pragma unroll
        for (int i = 0; i < 4; ++i) {
            uint4 hi, lo;
            cvt8(g[i][0], g[i][1], hi, lo);
            int eoff = (row0 + 16 * i) * STR + cg * 8;
            *(uint4*)(axs + eoff)        = hi;
            *(uint4*)(axs + LOFF + eoff) = lo;
        }
        __syncthreads();                 // tile visible

#pragma unroll
        for (int st = 0; st < BK / 32; ++st) {
            const int kl = st * 32;             // k offset within tile
            const int kc = kt * (BK / 32) + st; // global 32-K chunk index

            // ---- B fragments: raw 2x b128 from pre-swizzled Whi/Wlo (L2-hot)
            size_t boff = ((size_t)(kc * 2 + s) * NEXP + (size_t)(n0 + r)) * 16;
            Frag32B bh, bl;
            bh.u4[0] = ((const uint4*)(Whi + boff))[0];
            bh.u4[1] = ((const uint4*)(Whi + boff))[1];
            bl.u4[0] = ((const uint4*)(Wlo + boff))[0];
            bl.u4[1] = ((const uint4*)(Wlo + boff))[1];

            // ---- 4 M-tiles x 3 split terms, A fragments straight from LDS.
            // Contiguous A 16x32 layout: lane-half s -> elements 0..15 =
            // K kl+16s+0..15 (contiguous bf16).
#pragma unroll
            for (int m = 0; m < 4; ++m) {
                const int abase = (m * 16 + r) * STR + kl + s * 16;
                Frag32B ah, al;
                ah.u4[0] = *(const uint4*)(axs + abase);
                ah.u4[1] = *(const uint4*)(axs + abase + 8);
                al.u4[0] = *(const uint4*)(axs + LOFF + abase);
                al.u4[1] = *(const uint4*)(axs + LOFF + abase + 8);

                acc[m] = __builtin_amdgcn_wmma_f32_16x16x32_bf16(
                    false, ah.bf, false, bh.bf, (short)0, acc[m], false, false);
                acc[m] = __builtin_amdgcn_wmma_f32_16x16x32_bf16(
                    false, al.bf, false, bh.bf, (short)0, acc[m], false, false);
                acc[m] = __builtin_amdgcn_wmma_f32_16x16x32_bf16(
                    false, ah.bf, false, bl.bf, (short)0, acc[m], false, false);
            }
        }
    }

    // D layout: VGPR rr, lane -> M = rr + 8*s, N = r
#pragma unroll
    for (int m = 0; m < 4; ++m) {
#pragma unroll
        for (int rr = 0; rr < 8; ++rr) {
            int row = tokenBase + m * 16 + rr + 8 * s;
            logits[(size_t)row * NEXP + n0 + r] = acc[m][rr];
        }
    }
}

// ---------------------------------------------------------------------------
// Kernel 2: per-token routing. One wave per token; lane owns 4 contiguous
// experts so each expert-group (16 experts) is a quad of lanes.
// ---------------------------------------------------------------------------
__global__ __launch_bounds__(256) void router_topk_kernel(
    const float* __restrict__ logits,
    const float* __restrict__ bias,
    float* __restrict__ outIdx,
    float* __restrict__ outW,
    int T) {
    const int t    = blockIdx.x * 8 + (threadIdx.x >> 5);
    const int lane = threadIdx.x & 31;
    if (t >= T) return;

    float4 lg = ((const float4*)(logits + (size_t)t * NEXP))[lane];
    float4 bs = ((const float4*)bias)[lane];

    // raw sigmoid scores (weights) and bias-corrected scores (selection)
    float r0 = 1.0f / (1.0f + expf(-lg.x));
    float r1 = 1.0f / (1.0f + expf(-lg.y));
    float r2 = 1.0f / (1.0f + expf(-lg.z));
    float r3 = 1.0f / (1.0f + expf(-lg.w));
    float c0 = r0 + bs.x, c1 = r1 + bs.y, c2 = r2 + bs.z, c3 = r3 + bs.w;

    // in-lane top-2 of {c0..c3}
    float a = fmaxf(c0, c1), b = fminf(c0, c1);
    float c = fmaxf(c2, c3), d = fminf(c2, c3);
    float m1 = fmaxf(a, c);
    float m2 = fmaxf(fminf(a, c), fmaxf(b, d));
    // merge across the 4 lanes of this group (xor 1, xor 2 stay in-quad)
#pragma unroll
    for (int m = 1; m <= 2; m <<= 1) {
        float o1 = __shfl_xor(m1, m);
        float o2 = __shfl_xor(m2, m);
        float M1 = fmaxf(m1, o1);
        float M2 = fmaxf(fminf(m1, o1), fmaxf(m2, o2));
        m1 = M1; m2 = M2;
    }
    float gsum = m1 + m2;          // group score for group g = lane>>2

    // rank my group among the 8 group scores (ties -> lower index wins)
    const int g = lane >> 2;
    int cnt = 0;
#pragma unroll
    for (int i = 0; i < 8; ++i) {
        float gi = __shfl(gsum, i * 4);
        cnt += (gi > gsum) || (gi == gsum && i < g);
    }
    const bool sel = cnt < 4;      // TOPK_GROUP = 4

    // masked scores exactly like reference: where(mask, corrected, 0.0)
    float k0 = sel ? c0 : 0.0f;
    float k1 = sel ? c1 : 0.0f;
    float k2 = sel ? c2 : 0.0f;
    float k3 = sel ? c3 : 0.0f;

    const int base = lane * 4;
    float sum = 0.0f;
    int   myIdx = 0;
    float myW   = 0.0f;

#pragma unroll
    for (int k = 0; k < TOPK; ++k) {
        // in-lane argmax (ascending scan + strict > keeps smallest index)
        float bv = k0; int bi = base;     float br = r0;
        if (k1 > bv) { bv = k1; bi = base + 1; br = r1; }
        if (k2 > bv) { bv = k2; bi = base + 2; br = r2; }
        if (k3 > bv) { bv = k3; bi = base + 3; br = r3; }
        // wave argmax, tie-break by smaller index; carry raw weight
#pragma unroll
        for (int m = 1; m < 32; m <<= 1) {
            float ov = __shfl_xor(bv, m);
            int   oi = __shfl_xor(bi, m);
            float orw = __shfl_xor(br, m);
            bool take = (ov > bv) || (ov == bv && oi < bi);
            bv = take ? ov : bv;
            bi = take ? oi : bi;
            br = take ? orw : br;
        }
        sum += br;
        if (lane == k) { myIdx = bi; myW = br; }
        if ((bi >> 2) == lane) {          // owner retires the winner
            int jj = bi & 3;
            if      (jj == 0) k0 = -INFINITY;
            else if (jj == 1) k1 = -INFINITY;
            else if (jj == 2) k2 = -INFINITY;
            else              k3 = -INFINITY;
        }
    }

    float denom = sum + 1e-20f;           // NORM_TOPK_PROB, scale = 1.0
    if (lane < TOPK) {
        outIdx[(size_t)t * TOPK + lane] = (float)myIdx;
        outW [(size_t)t * TOPK + lane] = myW / denom;
    }
}

// ---------------------------------------------------------------------------
extern "C" void kernel_launch(void* const* d_in, const int* in_sizes, int n_in,
                              void* d_out, int out_size, void* d_ws, size_t ws_size,
                              hipStream_t stream) {
    const float* X    = (const float*)d_in[0];  // [4,8192,4096] fp32
    const float* W    = (const float*)d_in[1];  // [128,4096] fp32
    const float* bias = (const float*)d_in[2];  // [128] fp32
    const int T = in_sizes[0] / HIDDEN;         // 32768 tokens

    // workspace: Whi (1MB) | Wlo (1MB) | logits (T*128*4 = 16MB)
    char* ws = (char*)d_ws;
    unsigned short* whi = (unsigned short*)ws;
    unsigned short* wlo = (unsigned short*)(ws + (size_t)NEXP * HIDDEN * 2);
    float* logits       = (float*)        (ws + (size_t)NEXP * HIDDEN * 4);

    float* outIdx = (float*)d_out;               // first T*8: indices (as float)
    float* outW   = outIdx + (size_t)T * TOPK;   // next  T*8: weights

    const int pre_threads = (HIDDEN / 16) * NEXP;        // 32768
    preswz_kernel<<<pre_threads / 256, 256, 0, stream>>>(W, whi, wlo);
    router_gemm_kernel<<<T / MBLK, 256, 0, stream>>>(X, whi, wlo, logits);
    router_topk_kernel<<<(T + 7) / 8, 256, 0, stream>>>(logits, bias, outIdx, outW, T);
}
